// Attention_Layer_14903536517780
// MI455X (gfx1250) — matrix operand order repaired and checked by reading, not compiled, not run on hardware
//
#include <hip/hip_runtime.h>
#include <hip/hip_bf16.h>

typedef _Float16 v16h __attribute__((ext_vector_type(16)));
typedef _Float16 half8 __attribute__((ext_vector_type(8)));
typedef float    v8f  __attribute__((ext_vector_type(8)));
typedef float    vf2  __attribute__((ext_vector_type(2)));

#define NB   4096
#define TT   200
#define DD   64
#define MT   13            // ceil(208/16) row tiles
#define NEG_BIG (-4294967295.0f)

struct __align__(16) Smem {
  _Float16 info[208 * 128];   // 53248 B; overlaid by float h2[208*48] (39936 B) after GEMM1
  _Float16 w1t[80 * 256];     // 40960 B  (W1^T, f16)
  _Float16 h1[208 * 96];      // 39936 B  (zero-padded K for GEMM2)
  _Float16 w2t[48 * 96];      // 9216 B   (W2^T, f16, zero-padded)
  float qs[64];
  float b1[80];
  float b2[48];
  float wf[40];
  float bfv;
  float scores[208];
  float red[256];
};

// ---- pre-pass: transpose/convert weights once into d_ws ----
__global__ void prep_weights(const float* __restrict__ W1, const float* __restrict__ W2,
                             _Float16* __restrict__ w1t, _Float16* __restrict__ w2t) {
  const int gid = blockIdx.x * blockDim.x + threadIdx.x;
  const int stride = gridDim.x * blockDim.x;
  for (int j = gid; j < 80 * 256; j += stride) {
    const int n = j >> 8, f = j & 255;
    w1t[j] = (_Float16)W1[f * 80 + n];             // W1 is [256][80]
  }
  for (int j = gid; j < 48 * 96; j += stride) {
    const int n = j / 96, kk = j - n * 96;
    const float v = (n < 40 && kk < 80) ? W2[kk * 40 + n] : 0.0f;  // W2 is [80][40]
    w2t[j] = (_Float16)v;
  }
}

static __device__ inline v16h cat16(half8 lo, half8 hi) {
  return __builtin_shufflevector(lo, hi, 0,1,2,3,4,5,6,7,8,9,10,11,12,13,14,15);
}

__global__ void __launch_bounds__(256) din_attn(
    const float* __restrict__ qg, const float* __restrict__ kg,
    const float* __restrict__ vg, const int* __restrict__ maskg,
    const float* __restrict__ b1g, const float* __restrict__ b2g,
    const float* __restrict__ wfg, const float* __restrict__ bfg,
    const _Float16* __restrict__ w1tg, const _Float16* __restrict__ w2tg,
    float* __restrict__ outg) {
  __shared__ Smem sm;
  const int tid  = threadIdx.x;
  const int b    = blockIdx.x;
  const int lane = tid & 31;
  const int wave = tid >> 5;
  const int c    = lane & 15;   // WMMA column / A-row index
  const int h    = lane >> 4;   // lane half

  // ---- phase 0: stage weights/consts into LDS, zero h1 ----
  {
    const half8* s1 = (const half8*)w1tg; half8* d1 = (half8*)sm.w1t;
    for (int i = tid; i < (80 * 256) / 8; i += 256) d1[i] = s1[i];
    const half8* s2 = (const half8*)w2tg; half8* d2 = (half8*)sm.w2t;
    for (int i = tid; i < (48 * 96) / 8; i += 256) d2[i] = s2[i];
    if (tid < 64) sm.qs[tid] = qg[(size_t)b * 64 + tid];
    if (tid < 80) sm.b1[tid] = b1g[tid];
    if (tid < 48) sm.b2[tid] = (tid < 40) ? b2g[tid] : 0.0f;
    if (tid < 40) sm.wf[tid] = wfg[tid];
    if (tid == 0) sm.bfv = bfg[0];
    unsigned* hz = (unsigned*)sm.h1;
    for (int i = tid; i < (208 * 96) / 2; i += 256) hz[i] = 0u;
  }

  const v8f vzero = {};
  v8f acc[2][5];
  #pragma unroll
  for (int s = 0; s < 2; ++s)
    #pragma unroll
    for (int nt = 0; nt < 5; ++nt) acc[s][nt] = vzero;

  // ---- GEMM1: h1 = relu(info @ W1 + b1), K=256 done in two 128-wide passes ----
  for (int pass = 0; pass < 2; ++pass) {
    __syncthreads();   // phase0 done / previous pass readers done
    for (int pp = tid; pp < TT * 32; pp += 256) {     // build info half-chunk (f16)
      const int t = pp >> 5;
      const int cc = (pp & 31) * 2;
      const vf2 kv = *(const vf2*)(kg + (((size_t)b * TT + t) * DD + cc));
      const float q0 = sm.qs[cc], q1 = sm.qs[cc + 1];
      _Float16* row = sm.info + t * 128;
      if (pass == 0) {                                 // [q | k]
        row[cc]        = (_Float16)q0;     row[cc + 1]      = (_Float16)q1;
        row[64 + cc]   = (_Float16)kv[0];  row[64 + cc + 1] = (_Float16)kv[1];
      } else {                                         // [q-k | q*k]
        row[cc]        = (_Float16)(q0 - kv[0]); row[cc + 1]      = (_Float16)(q1 - kv[1]);
        row[64 + cc]   = (_Float16)(q0 * kv[0]); row[64 + cc + 1] = (_Float16)(q1 * kv[1]);
      }
    }
    for (int i = tid; i < 8 * 128; i += 256) sm.info[TT * 128 + i] = (_Float16)0.0f;  // pad rows
    __syncthreads();

    #pragma unroll
    for (int s = 0; s < 2; ++s) {
      const int mt = wave + s * 8;
      if (mt < MT) {                                   // wave-uniform: EXEC stays full
        const _Float16* arow = sm.info + (mt * 16 + c) * 128 + 8 * h;
        #pragma unroll
        for (int ksl = 0; ksl < 4; ++ksl) {
          const half8 alo = *(const half8*)(arow + ksl * 32);
          const half8 ahi = *(const half8*)(arow + ksl * 32 + 16);
          const v16h afrag = cat16(alo, ahi);
          const int ks = pass * 4 + ksl;
          #pragma unroll
          for (int nt = 0; nt < 5; ++nt) {
            const _Float16* bp = sm.w1t + (nt * 16 + c) * 256 + ks * 32 + 8 * h;
            const v16h bfrag = cat16(*(const half8*)bp, *(const half8*)(bp + 16));
            acc[s][nt] = __builtin_amdgcn_wmma_f32_16x16x32_f16(
                false, afrag, false, bfrag, (short)0, acc[s][nt], false, false);
          }
        }
      }
    }
  }

  // store h1 (+bias, relu) as f16
  #pragma unroll
  for (int s = 0; s < 2; ++s) {
    const int mt = wave + s * 8;
    if (mt < MT) {
      #pragma unroll
      for (int nt = 0; nt < 5; ++nt) {
        const int n = nt * 16 + c;
        const float bv = sm.b1[n];
        #pragma unroll
        for (int m = 0; m < 8; ++m) {
          const int t = mt * 16 + 8 * h + m;
          sm.h1[t * 96 + n] = (_Float16)fmaxf(acc[s][nt][m] + bv, 0.0f);
        }
      }
    }
  }
  __syncthreads();

  // ---- GEMM2: h2 = relu(h1 @ W2 + b2); h2 (f32) overlays dead info region ----
  float* h2 = (float*)sm.info;
  v8f acc2[2][3];
  #pragma unroll
  for (int s = 0; s < 2; ++s)
    #pragma unroll
    for (int nt = 0; nt < 3; ++nt) acc2[s][nt] = vzero;

  #pragma unroll
  for (int s = 0; s < 2; ++s) {
    const int mt = wave + s * 8;
    if (mt < MT) {
      const _Float16* arow = sm.h1 + (mt * 16 + c) * 96 + 8 * h;
      #pragma unroll
      for (int ksl = 0; ksl < 3; ++ksl) {
        const half8 alo = *(const half8*)(arow + ksl * 32);
        const half8 ahi = *(const half8*)(arow + ksl * 32 + 16);
        const v16h afrag = cat16(alo, ahi);
        #pragma unroll
        for (int nt = 0; nt < 3; ++nt) {
          const _Float16* bp = sm.w2t + (nt * 16 + c) * 96 + ksl * 32 + 8 * h;
          const v16h bfrag = cat16(*(const half8*)bp, *(const half8*)(bp + 16));
          acc2[s][nt] = __builtin_amdgcn_wmma_f32_16x16x32_f16(
              false, afrag, false, bfrag, (short)0, acc2[s][nt], false, false);
        }
      }
      #pragma unroll
      for (int nt = 0; nt < 3; ++nt) {
        const int n = nt * 16 + c;
        const float bv = sm.b2[n];
        #pragma unroll
        for (int m = 0; m < 8; ++m) {
          const int t = mt * 16 + 8 * h + m;
          h2[t * 48 + n] = fmaxf(acc2[s][nt][m] + bv, 0.0f);
        }
      }
    }
  }
  __syncthreads();

  // ---- scores = h2 . Wf + bf, mask ----
  if (tid < TT) {
    float sv = sm.bfv;
    const float* hrow = h2 + tid * 48;
    #pragma unroll 8
    for (int g = 0; g < 40; ++g) sv += hrow[g] * sm.wf[g];
    sv = (maskg[(size_t)b * TT + tid] == 0) ? NEG_BIG : sv;
    sm.scores[tid] = sv;
  }
  __syncthreads();

  // ---- softmax over T ----
  sm.red[tid] = (tid < TT) ? sm.scores[tid] : -3.0e38f;
  __syncthreads();
  for (int off = 128; off > 0; off >>= 1) {
    if (tid < off) sm.red[tid] = fmaxf(sm.red[tid], sm.red[tid + off]);
    __syncthreads();
  }
  const float mx = sm.red[0];
  __syncthreads();
  const float e = (tid < TT) ? __expf(sm.scores[tid] - mx) : 0.0f;
  sm.red[tid] = e;
  __syncthreads();
  for (int off = 128; off > 0; off >>= 1) {
    if (tid < off) sm.red[tid] += sm.red[tid + off];
    __syncthreads();
  }
  const float inv = 1.0f / sm.red[0];
  __syncthreads();
  if (tid < TT) sm.scores[tid] = e * inv;
  __syncthreads();

  // ---- out[b,:] = attn . v  (coalesced v reads) ----
  const int col  = tid & 63;
  const int part = tid >> 6;                 // 4 chunks of 50 timesteps
  float a = 0.0f;
  const float* vb = vg + ((size_t)b * TT + part * 50) * DD + col;
  for (int t = 0; t < 50; ++t) a += sm.scores[part * 50 + t] * vb[t * DD];
  sm.red[tid] = a;
  __syncthreads();
  if (tid < 64)
    outg[(size_t)b * 64 + tid] =
        sm.red[tid] + sm.red[tid + 64] + sm.red[tid + 128] + sm.red[tid + 192];
}

extern "C" void kernel_launch(void* const* d_in, const int* in_sizes, int n_in,
                              void* d_out, int out_size, void* d_ws, size_t ws_size,
                              hipStream_t stream) {
  const float* q    = (const float*)d_in[0];
  const float* k    = (const float*)d_in[1];
  const float* v    = (const float*)d_in[2];
  const int*   mask = (const int*)d_in[3];
  const float* W1   = (const float*)d_in[4];
  const float* b1   = (const float*)d_in[5];
  const float* W2   = (const float*)d_in[6];
  const float* b2   = (const float*)d_in[7];
  const float* Wf   = (const float*)d_in[8];
  const float* bf   = (const float*)d_in[9];
  float* out = (float*)d_out;

  _Float16* w1t = (_Float16*)d_ws;           // 80*256 f16 = 40960 B
  _Float16* w2t = w1t + 80 * 256;            // 48*96  f16 =  9216 B

  prep_weights<<<64, 256, 0, stream>>>(W1, W2, w1t, w2t);
  din_attn<<<NB, 256, 0, stream>>>(q, k, v, mask, b1, b2, Wf, bf, w1t, w2t, out);
}
